// TemporalGCN_51883204935836
// MI455X (gfx1250) — matrix operand order repaired and checked by reading, not compiled, not run on hardware
//
#include <hip/hip_runtime.h>
#include <hip/hip_bf16.h>

// ---------------------------------------------------------------------------
// TemporalGCN for MI455X (gfx1250, wave32).
// - Edge aggregation dominates (~1.1 GB L2-resident traffic per GCN layer;
//   xw + agg = 64 MB << 192 MB L2) -> wave-per-edge f32 atomics at L2 speed.
// - GEMMs: v_wmma_f32_16x16x32_bf16, K templated (fully unrolled), A tile in
//   LDS read as b128, B pre-packed into fragment layout for coalesced b128.
// ---------------------------------------------------------------------------

#define B_   64
#define C_   64
#define T_   2048
#define HID_ 256
#define OUT_ 10
#define NEWT 512            // T/4
#define N_   (B_ * NEWT)    // 32768 nodes

typedef __attribute__((ext_vector_type(16))) __bf16 v16bf;
typedef __attribute__((ext_vector_type(8)))  float  v8f;

union FragAB { v16bf v; unsigned short u[16]; uint4 q[2]; };

static __device__ __forceinline__ unsigned short f2bf(float f) {
    unsigned int u = __float_as_uint(f);
    u += 0x7FFFu + ((u >> 16) & 1u);   // round-to-nearest-even
    return (unsigned short)(u >> 16);
}

// ---------------------------------------------------------------------------
// Conv1d(64->16, K=5, pad=2) + ReLU + maxpool2.  Block: 16 co x 16 t, 256 thr.
// ---------------------------------------------------------------------------
__global__ void conv1_relu_pool(const float* __restrict__ x,
                                const float* __restrict__ w,
                                const float* __restrict__ bias,
                                float* __restrict__ h1) {
    __shared__ float ws[16 * 64 * 5];
    __shared__ float bs[16];
    for (int i = threadIdx.x; i < 16 * 64 * 5; i += 256) ws[i] = w[i];
    if (threadIdx.x < 16) bs[threadIdx.x] = bias[threadIdx.x];
    __syncthreads();

    int co   = threadIdx.x & 15;
    int tl   = threadIdx.x >> 4;          // 0..15
    int tblk = blockIdx.x & 63;           // 1024/16 tiles
    int bb   = blockIdx.x >> 6;           // batch
    int t    = tblk * 16 + tl;            // pooled output position, 0..1023
    const float* xb = x + (size_t)bb * C_ * T_;

    float r = 0.0f;
    #pragma unroll
    for (int p = 0; p < 2; ++p) {
        int tc = 2 * t + p;
        float acc = bs[co];
        for (int ci = 0; ci < 64; ++ci) {
            const float* xr = xb + (size_t)ci * T_;
            const float* wr = ws + (co * 64 + ci) * 5;
            #pragma unroll
            for (int k = 0; k < 5; ++k) {
                int tt = tc + k - 2;
                if (tt >= 0 && tt < T_) acc = fmaf(xr[tt], wr[k], acc);
            }
        }
        acc = fmaxf(acc, 0.0f);
        r = p ? fmaxf(r, acc) : acc;
    }
    h1[((size_t)bb * 16 + co) * 1024 + t] = r;
}

// ---------------------------------------------------------------------------
// Conv1d(16->32, K=5, pad=2) + ReLU + maxpool2 -> node-major bf16 [N,32].
// ---------------------------------------------------------------------------
__global__ void conv2_relu_pool(const float* __restrict__ h1,
                                const float* __restrict__ w,
                                const float* __restrict__ bias,
                                unsigned short* __restrict__ hn) {
    __shared__ float ws[32 * 16 * 5];
    __shared__ float bs[32];
    for (int i = threadIdx.x; i < 32 * 16 * 5; i += 256) ws[i] = w[i];
    if (threadIdx.x < 32) bs[threadIdx.x] = bias[threadIdx.x];
    __syncthreads();

    int co   = threadIdx.x & 31;
    int tl   = threadIdx.x >> 5;          // 0..7
    int tblk = blockIdx.x & 63;           // 512/8 tiles
    int bb   = blockIdx.x >> 6;
    int t    = tblk * 8 + tl;             // 0..511
    const float* hb = h1 + (size_t)bb * 16 * 1024;

    float r = 0.0f;
    #pragma unroll
    for (int p = 0; p < 2; ++p) {
        int tc = 2 * t + p;
        float acc = bs[co];
        for (int ci = 0; ci < 16; ++ci) {
            const float* xr = hb + (size_t)ci * 1024;
            const float* wr = ws + (co * 16 + ci) * 5;
            #pragma unroll
            for (int k = 0; k < 5; ++k) {
                int tt = tc + k - 2;
                if (tt >= 0 && tt < 1024) acc = fmaf(xr[tt], wr[k], acc);
            }
        }
        acc = fmaxf(acc, 0.0f);
        r = p ? fmaxf(r, acc) : acc;
    }
    int n = bb * NEWT + t;
    hn[(size_t)n * 32 + co] = f2bf(r);
}

// ---------------------------------------------------------------------------
// f32 -> bf16 conversion (weights)
// ---------------------------------------------------------------------------
__global__ void cvt_f32_bf16(const float* __restrict__ in,
                             unsigned short* __restrict__ out, int n) {
    int i = blockIdx.x * blockDim.x + threadIdx.x;
    if (i < n) out[i] = f2bf(in[i]);
}

// ---------------------------------------------------------------------------
// Pack B (bf16, row-major [K][256]) into WMMA-fragment-major layout:
//   Bp[((kt*16 + nt)*32 + lane)*16 + i] = B[kt*32 + (lane>=16?16:0) + i][nt*16 + (lane&15)]
// so each lane's 32-byte fragment is contiguous -> two global_load_b128.
// ---------------------------------------------------------------------------
__global__ void pack_b_frag(const unsigned short* __restrict__ B,
                            unsigned short* __restrict__ Bp, int K) {
    int idx = blockIdx.x * blockDim.x + threadIdx.x;   // K*256 total
    if (idx >= K * HID_) return;
    int i    = idx & 15;
    int lane = (idx >> 4) & 31;
    int nt   = (idx >> 9) & 15;
    int kt   = idx >> 13;
    int krow = kt * 32 + ((lane >= 16) ? 16 : 0) + i;
    int col  = nt * 16 + (lane & 15);
    Bp[idx] = B[(size_t)krow * HID_ + col];
}

// ---------------------------------------------------------------------------
// Degree / D^-1/2
// ---------------------------------------------------------------------------
__global__ void deg_init(float* __restrict__ deg) {
    int i = blockIdx.x * blockDim.x + threadIdx.x;
    if (i < N_) deg[i] = 1.0f;            // self-loop
}
__global__ void deg_accum(const int* __restrict__ dst, float* __restrict__ deg, int E) {
    int i = blockIdx.x * blockDim.x + threadIdx.x;
    if (i < E) atomicAdd(&deg[dst[i]], 1.0f);
}
__global__ void deg_rsqrt(float* __restrict__ deg) {
    int i = blockIdx.x * blockDim.x + threadIdx.x;
    if (i < N_) deg[i] = rsqrtf(deg[i]);
}

// ---------------------------------------------------------------------------
// WMMA bf16 GEMM:  C[M,256] = A[M,K] @ B[K,256], f32 accumulate.
// Block = 256 thr = 8 waves. A tile (16 x K) staged in LDS (b128 in/out).
// Each wave owns a 16x32 output strip (two N-tiles share one A fragment,
// 2 WMMAs per K-step). B is fragment-packed (Bp) -> coalesced b128 loads.
// K is a template constant so the kt-loop fully unrolls.
// ---------------------------------------------------------------------------
template <int K>
__global__ void gemm_bf16_wmma(const unsigned short* __restrict__ A,
                               const unsigned short* __restrict__ Bp,
                               float* __restrict__ C) {
    __shared__ unsigned short As[16 * K];
    const int m0 = blockIdx.x * 16;

    // cooperative A-tile stage, 128-bit granules
    {
        const uint4* src = (const uint4*)(A + (size_t)m0 * K);
        uint4* dst = (uint4*)As;
        const int n128 = (16 * K) / 8;
        for (int i = threadIdx.x; i < n128; i += 256) dst[i] = src[i];
    }
    __syncthreads();

    const int lane = threadIdx.x & 31;
    const int wave = threadIdx.x >> 5;
    const int nt0  = wave * 2;            // two N-tiles per wave
    const bool hi  = lane >= 16;
    const int arow = hi ? lane - 16 : lane;

    v8f acc0 = {};
    v8f acc1 = {};
    const uint4* arow128 = (const uint4*)(As + arow * K);

    #pragma unroll
    for (int kt = 0; kt < K / 32; ++kt) {
        const int kk = kt * 32;
        // A fragment (matches packed B): lanes 0-15 -> K {kk..+15};
        //                                lanes 16-31 -> K {kk+16..+31}
        const int ka = kk + (hi ? 16 : 0);       // multiple of 8 -> 16B aligned
        const int kb = ka + 8;
        FragAB fa;
        fa.q[0] = arow128[ka >> 3];
        fa.q[1] = arow128[kb >> 3];
        // B fragments: 32 contiguous bytes per lane from packed layout
        const uint4* b0 = (const uint4*)(Bp + ((size_t)((kt * 16 + nt0) * 32 + lane) * 16));
        const uint4* b1 = (const uint4*)(Bp + ((size_t)((kt * 16 + nt0 + 1) * 32 + lane) * 16));
        FragAB fb0, fb1;
        fb0.q[0] = b0[0]; fb0.q[1] = b0[1];
        fb1.q[0] = b1[0]; fb1.q[1] = b1[1];

        acc0 = __builtin_amdgcn_wmma_f32_16x16x32_bf16(
            false, fa.v, false, fb0.v, (short)0, acc0, false, false);
        acc1 = __builtin_amdgcn_wmma_f32_16x16x32_bf16(
            false, fa.v, false, fb1.v, (short)0, acc1, false, false);
    }

    // C/D layout: VGPR i -> row m0 + i (+8 for high lanes), col = lane%16
    const int cc    = hi ? lane - 16 : lane;
    const int rbase = hi ? 8 : 0;
    float* c0 = C + (size_t)(m0 + rbase) * HID_ + nt0 * 16 + cc;
    #pragma unroll
    for (int i = 0; i < 8; ++i) {
        c0[(size_t)i * HID_]      = acc0[i];
        c0[(size_t)i * HID_ + 16] = acc1[i];
    }
}

// ---------------------------------------------------------------------------
// agg[n,c] = xw[n,c] * dinv[n]^2 + bias[c]   (self-loop message + bias)
// ---------------------------------------------------------------------------
__global__ void agg_init(const float* __restrict__ xw, const float* __restrict__ dinv,
                         const float* __restrict__ bias, float* __restrict__ agg) {
    int idx = blockIdx.x * blockDim.x + threadIdx.x;   // N_*HID_ total
    int n = idx >> 8, c = idx & 255;
    float dv = dinv[n];
    agg[idx] = xw[idx] * dv * dv + bias[c];
}

// ---------------------------------------------------------------------------
// Edge scatter: one wave per edge; 256 features as 64x float4; f32 atomics.
// xw + agg are L2-resident, so this runs at L2 bandwidth.
// ---------------------------------------------------------------------------
__global__ void edge_aggregate(const float* __restrict__ xw,
                               const int* __restrict__ src,
                               const int* __restrict__ dst,
                               const float* __restrict__ dinv,
                               float* __restrict__ agg, int E) {
    int wid  = (blockIdx.x * blockDim.x + threadIdx.x) >> 5;
    int lane = threadIdx.x & 31;
    if (wid >= E) return;
    int s = src[wid], d = dst[wid];
    if (wid + 8 < E)   // warm L2 for a future gather row
        __builtin_prefetch(xw + (size_t)src[wid + 8] * HID_, 0, 1);
    float w = dinv[s] * dinv[d];
    const float4* ps = (const float4*)(xw + (size_t)s * HID_);
    float* pd = agg + (size_t)d * HID_;
    #pragma unroll
    for (int j = 0; j < 2; ++j) {
        int c4 = lane + j * 32;
        float4 v = ps[c4];
        atomicAdd(pd + c4 * 4 + 0, v.x * w);
        atomicAdd(pd + c4 * 4 + 1, v.y * w);
        atomicAdd(pd + c4 * 4 + 2, v.z * w);
        atomicAdd(pd + c4 * 4 + 3, v.w * w);
    }
}

__global__ void relu_to_bf16(const float* __restrict__ in, unsigned short* __restrict__ out) {
    int i = blockIdx.x * blockDim.x + threadIdx.x;
    out[i] = f2bf(fmaxf(in[i], 0.0f));
}
__global__ void relu_inplace(float* __restrict__ a) {
    int i = blockIdx.x * blockDim.x + threadIdx.x;
    a[i] = fmaxf(a[i], 0.0f);
}

// ---------------------------------------------------------------------------
// Mean over 512 temporal nodes per batch, then [64,256]@[256,10]+b
// ---------------------------------------------------------------------------
__global__ void pool_classify(const float* __restrict__ h,
                              const float* __restrict__ cw,
                              const float* __restrict__ cb,
                              float* __restrict__ out) {
    __shared__ float pooled[HID_];
    int bb = blockIdx.x, c = threadIdx.x;
    float acc = 0.0f;
    const float* base = h + ((size_t)bb * NEWT) * HID_ + c;
    for (int i = 0; i < NEWT; ++i) acc += base[(size_t)i * HID_];
    pooled[c] = acc * (1.0f / (float)NEWT);
    __syncthreads();
    if (c < OUT_) {
        float o = cb[c];
        for (int k = 0; k < HID_; ++k) o = fmaf(pooled[k], cw[k * OUT_ + c], o);
        out[bb * OUT_ + c] = o;
    }
}

// ---------------------------------------------------------------------------
extern "C" void kernel_launch(void* const* d_in, const int* in_sizes, int n_in,
                              void* d_out, int out_size, void* d_ws, size_t ws_size,
                              hipStream_t stream) {
    const float* x       = (const float*)d_in[0];
    const int*   eidx    = (const int*)d_in[1];
    const float* conv1_w = (const float*)d_in[2];
    const float* conv1_b = (const float*)d_in[3];
    const float* conv2_w = (const float*)d_in[4];
    const float* conv2_b = (const float*)d_in[5];
    const float* gcn1_w  = (const float*)d_in[6];
    const float* gcn1_b  = (const float*)d_in[7];
    const float* gcn2_w  = (const float*)d_in[8];
    const float* gcn2_b  = (const float*)d_in[9];
    const float* cls_w   = (const float*)d_in[10];
    const float* cls_b   = (const float*)d_in[11];
    float* out = (float*)d_out;

    const int E = in_sizes[1] / 2;
    const int* src = eidx;
    const int* dst = eidx + E;

    // workspace carve-up (256B aligned)
    char* ws = (char*)d_ws;
    size_t off = 0;
    auto take = [&](size_t bytes) { char* p = ws + off; off = (off + bytes + 255) & ~(size_t)255; return p; };
    float*          h1   = (float*)take((size_t)B_ * 16 * 1024 * 4);       // 4 MB
    unsigned short* hn   = (unsigned short*)take((size_t)N_ * 32 * 2);     // 2 MB
    unsigned short* w1b  = (unsigned short*)take((size_t)32 * HID_ * 2);
    unsigned short* w2b  = (unsigned short*)take((size_t)HID_ * HID_ * 2);
    unsigned short* w1p  = (unsigned short*)take((size_t)32 * HID_ * 2);   // packed
    unsigned short* w2p  = (unsigned short*)take((size_t)HID_ * HID_ * 2); // packed
    float*          dinv = (float*)take((size_t)N_ * 4);
    float*          xw   = (float*)take((size_t)N_ * HID_ * 4);            // 32 MB
    float*          agg  = (float*)take((size_t)N_ * HID_ * 4);            // 32 MB
    unsigned short* hb   = (unsigned short*)take((size_t)N_ * HID_ * 2);   // 16 MB
    (void)ws_size; (void)n_in; (void)out_size;

    // stage 1: temporal convs
    conv1_relu_pool<<<4096, 256, 0, stream>>>(x, conv1_w, conv1_b, h1);
    conv2_relu_pool<<<4096, 256, 0, stream>>>(h1, conv2_w, conv2_b, hn);

    // weights -> bf16 -> fragment-packed
    cvt_f32_bf16<<<(32 * HID_ + 255) / 256, 256, 0, stream>>>(gcn1_w, w1b, 32 * HID_);
    cvt_f32_bf16<<<(HID_ * HID_ + 255) / 256, 256, 0, stream>>>(gcn2_w, w2b, HID_ * HID_);
    pack_b_frag<<<(32 * HID_ + 255) / 256, 256, 0, stream>>>(w1b, w1p, 32);
    pack_b_frag<<<(HID_ * HID_ + 255) / 256, 256, 0, stream>>>(w2b, w2p, HID_);

    // symmetric normalization
    deg_init<<<N_ / 256, 256, 0, stream>>>(dinv);
    deg_accum<<<(E + 255) / 256, 256, 0, stream>>>(dst, dinv, E);
    deg_rsqrt<<<N_ / 256, 256, 0, stream>>>(dinv);

    const int nc_blocks   = (N_ * HID_) / 256;
    const int edge_blocks = (E + 7) / 8;   // one wave per edge, 8 waves/block

    // GCN layer 1
    gemm_bf16_wmma<32><<<N_ / 16, 256, 0, stream>>>(hn, w1p, xw);
    agg_init<<<nc_blocks, 256, 0, stream>>>(xw, dinv, gcn1_b, agg);
    edge_aggregate<<<edge_blocks, 256, 0, stream>>>(xw, src, dst, dinv, agg, E);
    relu_to_bf16<<<nc_blocks, 256, 0, stream>>>(agg, hb);

    // GCN layer 2
    gemm_bf16_wmma<HID_><<<N_ / 16, 256, 0, stream>>>(hb, w2p, xw);
    agg_init<<<nc_blocks, 256, 0, stream>>>(xw, dinv, gcn2_b, agg);
    edge_aggregate<<<edge_blocks, 256, 0, stream>>>(xw, src, dst, dinv, agg, E);
    relu_inplace<<<nc_blocks, 256, 0, stream>>>(agg);

    // pool + classifier
    pool_classify<<<B_, HID_, 0, stream>>>(agg, cls_w, cls_b, out);
}
